// MultiHeadAttention_2774548873672
// MI455X (gfx1250) — matrix operand order repaired and checked by reading, not compiled, not run on hardware
//
#include <hip/hip_runtime.h>
#include <hip/hip_bf16.h>

typedef __attribute__((ext_vector_type(16))) __bf16 v16bf;
typedef __attribute__((ext_vector_type(8)))  float  v8f;
typedef __attribute__((ext_vector_type(2)))  __bf16 bf16x2;
typedef __attribute__((ext_vector_type(4)))  unsigned int u32x4;
typedef __attribute__((ext_vector_type(8)))  int          i32x8;
typedef __attribute__((ext_vector_type(4)))  int          i32x4;
typedef unsigned int u32;

#define SEQ 4096
#define DIN 768
#define NH  12
#define HD  64
#define HSZ (NH*SEQ*HD)   /* elements per projection matrix */

// ---------------- conversion kernels ----------------

__global__ void cvt_x_kernel(const float* __restrict__ x, __bf16* __restrict__ xb) {
    int i = blockIdx.x * 256 + threadIdx.x;
    xb[i] = (__bf16)x[i];
}

// Wt[mat][o][k] = W[mat][k][o]  (transpose so WMMA B-operand K-pairs are contiguous)
__global__ void cvt_w_kernel(const float* __restrict__ Wq, const float* __restrict__ Wk,
                             const float* __restrict__ Wv, __bf16* __restrict__ Wt) {
    int i = blockIdx.x * 256 + threadIdx.x;   // = o*768 + k
    int o = i / DIN;
    int k = i - o * DIN;
    const float* W = (blockIdx.y == 0) ? Wq : (blockIdx.y == 1) ? Wk : Wv;
    Wt[(size_t)blockIdx.y * DIN * DIN + i] = (__bf16)W[k * DIN + o];
}

// ---------------- projection GEMM (x @ W) via WMMA bf16 ----------------
// Block = 8 waves computes a 16-row x 128-col strip. The 16x768 A strip of x is
// staged once into LDS via the Tensor Data Mover and shared by all 8 waves.
// Q,K stored [h][s][64]; V stored transposed [h][64][s].

__global__ void proj_kernel(const __bf16* __restrict__ xb, const __bf16* __restrict__ Wt,
                            __bf16* __restrict__ qkv) {
    __shared__ __bf16 sA[16 * DIN];           // 24 KB A strip
    int wave = threadIdx.x >> 5;
    int lane = threadIdx.x & 31;
    int rm = blockIdx.x / (DIN / 128);        // row tile   [0,256)
    int cs = blockIdx.x - rm * (DIN / 128);   // col strip  [0,6)
    int tn = cs * 8 + wave;                   // this wave's 16-col tile
    int mat = blockIdx.y;

    // ---- stage A strip (16 x 768 bf16) into LDS ----
#if __has_builtin(__builtin_amdgcn_tensor_load_to_lds) && __has_builtin(__builtin_amdgcn_s_wait_tensorcnt)
    if (threadIdx.x < 32) {                   // wave 0 issues one TDM descriptor
        unsigned long long ga = (unsigned long long)(uintptr_t)(xb + (size_t)rm * 16 * DIN);
        u32 ldsoff = (u32)(uintptr_t)&sA[0];  // flat addr[31:0] == LDS offset
        u32x4 g0; i32x8 g1; i32x4 g2 = {0,0,0,0}, g3 = {0,0,0,0};
        // D# group0: count=1 | lds_addr | global_addr[56:0] | type=2
        g0[0] = 1u;
        g0[1] = ldsoff;
        g0[2] = (u32)ga;
        g0[3] = (u32)((ga >> 32) & 0x1FFFFFFull) | 0x80000000u;
        // D# group1: data_size=2B; tensor 768 x 4096; tile 768 x 16; dim0_stride=768
        g1[0] = (1 << 16);                             // data_size=1 (2 bytes)
        g1[1] = (int)((DIN & 0xFFFF) << 16);           // tensor_dim0 lo16
        g1[2] = (int)(((DIN >> 16) & 0xFFFF) | ((SEQ & 0xFFFF) << 16)); // dim0 hi | dim1 lo
        g1[3] = (int)(((SEQ >> 16) & 0xFFFF) | ((DIN & 0xFFFF) << 16)); // dim1 hi | tile_dim0
        g1[4] = 16;                                    // tile_dim1 = 16 rows
        g1[5] = DIN;                                   // tensor_dim0_stride lo32
        g1[6] = 0;                                     // dim0_stride hi | dim1_stride lo
        g1[7] = 0;
#if defined(__clang_major__) && (__clang_major__ >= 23)
        i32x8 g4 = {0,0,0,0,0,0,0,0};
        __builtin_amdgcn_tensor_load_to_lds(g0, g1, g2, g3, g4, 0);
#else
        __builtin_amdgcn_tensor_load_to_lds(g0, g1, g2, g3, 0);
#endif
        __builtin_amdgcn_s_wait_tensorcnt(0);
    }
    __syncthreads();
#else
    {   // fallback: cooperative 128-bit copy
        const uint4* src = (const uint4*)(xb + (size_t)rm * 16 * DIN);
        uint4* dst = (uint4*)sA;
        for (int i = threadIdx.x; i < 16 * DIN * 2 / 16; i += 256) dst[i] = src[i];
        __syncthreads();
    }
#endif

    const __bf16* W   = Wt  + (size_t)mat * DIN * DIN;
    __bf16*       dst = qkv + (size_t)mat * HSZ;

    int nlo = lane & 15, half = lane >> 4;
    const __bf16* arow = sA + (size_t)nlo * DIN;               // A from LDS: lane = row
    const __bf16* brow = W  + (size_t)(tn * 16 + nlo) * DIN;   // B: lane = column (Wt row)

    v8f c = {};
    for (int ks = 0; ks < DIN; ks += 32) {
        v16bf a, b;
#pragma unroll
        for (int v = 0; v < 8; ++v) {
            int ka = ((v & 4) << 2) + half * 8 + (v & 3) * 2;     // A 16x32 layout
            bf16x2 pa = __builtin_bit_cast(bf16x2, *(const unsigned int*)(arow + ks + ka));
            a[2 * v] = pa[0]; a[2 * v + 1] = pa[1];
            int kb = ((v & 4) << 2) + half * 8 + (v & 3) * 2;     // B 32x16 layout (same k order as A)
            bf16x2 pb = __builtin_bit_cast(bf16x2, *(const unsigned int*)(brow + ks + kb));
            b[2 * v] = pb[0]; b[2 * v + 1] = pb[1];
        }
        c = __builtin_amdgcn_wmma_f32_16x16x32_bf16(false, a, false, b, (short)0, c, false, false);
    }

    int n = tn * 16 + nlo;          // output feature
    int head = n >> 6, d = n & 63;
#pragma unroll
    for (int v = 0; v < 8; ++v) {
        int r = rm * 16 + v + half * 8;   // C/D layout: row = v + 8*half
        size_t idx = (mat == 2)
                   ? (size_t)head * HD * SEQ + (size_t)d * SEQ + r     // V transposed
                   : (size_t)head * SEQ * HD + (size_t)r * HD + d;     // Q,K row-major
        dst[idx] = (__bf16)c[v];
    }
}

// ---------------- flash attention: 1 wave = 16 query rows ----------------

__global__ void attn_kernel(const __bf16* __restrict__ qkv, float* __restrict__ out) {
    __shared__ __bf16 plds[8][16][32];
    int wave = threadIdx.x >> 5;
    int lane = threadIdx.x & 31;
    int h = blockIdx.y;
    int qbase = (blockIdx.x * 8 + wave) * 16;
    const __bf16* Q  = qkv + (size_t)h * SEQ * HD;
    const __bf16* K  = qkv + (size_t)HSZ + (size_t)h * SEQ * HD;
    const __bf16* Vt = qkv + (size_t)2 * HSZ + (size_t)h * HD * SEQ;  // [64][4096]
    int nlo = lane & 15, half = lane >> 4;

    // Q as two A-fragments (head dim 64 = 2 x K32)
    v16bf qa0, qa1;
    {
        const __bf16* qrow = Q + (size_t)(qbase + nlo) * HD;
#pragma unroll
        for (int v = 0; v < 8; ++v) {
            int ka = ((v & 4) << 2) + half * 8 + (v & 3) * 2;
            bf16x2 p0 = __builtin_bit_cast(bf16x2, *(const unsigned int*)(qrow + ka));
            qa0[2 * v] = p0[0]; qa0[2 * v + 1] = p0[1];
            bf16x2 p1 = __builtin_bit_cast(bf16x2, *(const unsigned int*)(qrow + 32 + ka));
            qa1[2 * v] = p1[0]; qa1[2 * v + 1] = p1[1];
        }
    }

    v8f o0 = {}, o1 = {}, o2 = {}, o3 = {};
    float mrow[8], lrow[8];
#pragma unroll
    for (int v = 0; v < 8; ++v) { mrow[v] = -1e30f; lrow[v] = 0.f; }

    auto loadB = [&](const __bf16* base) -> v16bf {
        v16bf b;
#pragma unroll
        for (int v = 0; v < 8; ++v) {
            int kb = ((v & 4) << 2) + half * 8 + (v & 3) * 2;
            bf16x2 p = __builtin_bit_cast(bf16x2, *(const unsigned int*)(base + kb));
            b[2 * v] = p[0]; b[2 * v + 1] = p[1];
        }
        return b;
    };

    int ntiles = (qbase + 15) / 32 + 1;     // causal: cover keys <= qbase+15
    for (int jt = 0; jt < ntiles; ++jt) {
        int jb = jt * 32;

        // S(16x32) = Q(16x64) . K_tile^T : 4 WMMAs (2 key sub-tiles x 2 K-chunks)
        v8f s0 = {}, s1 = {};
        {
            const __bf16* k0 = K + (size_t)(jb + nlo) * HD;       // lane = key column
            const __bf16* k1 = K + (size_t)(jb + 16 + nlo) * HD;
            s0 = __builtin_amdgcn_wmma_f32_16x16x32_bf16(false, qa0, false, loadB(k0),      (short)0, s0, false, false);
            s0 = __builtin_amdgcn_wmma_f32_16x16x32_bf16(false, qa1, false, loadB(k0 + 32), (short)0, s0, false, false);
            s1 = __builtin_amdgcn_wmma_f32_16x16x32_bf16(false, qa0, false, loadB(k1),      (short)0, s1, false, false);
            s1 = __builtin_amdgcn_wmma_f32_16x16x32_bf16(false, qa1, false, loadB(k1 + 32), (short)0, s1, false, false);
        }

        // online softmax over the 32 keys of this tile
#pragma unroll
        for (int v = 0; v < 8; ++v) {
            int row = qbase + v + half * 8;
            float x0 = s0[v] * 0.125f;                 // 1/sqrt(64)
            float x1 = s1[v] * 0.125f;
            if (jb + nlo > row)      x0 = -1e30f;      // causal mask
            if (jb + 16 + nlo > row) x1 = -1e30f;
            float mx = fmaxf(x0, x1);
#pragma unroll
            for (int off = 8; off; off >>= 1) mx = fmaxf(mx, __shfl_xor(mx, off, 32));
            float mnew  = fmaxf(mrow[v], mx);
            float scale = __expf(mrow[v] - mnew);
            float p0 = __expf(x0 - mnew);
            float p1 = __expf(x1 - mnew);
            float rs = p0 + p1;
#pragma unroll
            for (int off = 8; off; off >>= 1) rs += __shfl_xor(rs, off, 32);
            lrow[v] = lrow[v] * scale + rs;
            mrow[v] = mnew;
            o0[v] *= scale; o1[v] *= scale; o2[v] *= scale; o3[v] *= scale;
            plds[wave][v + half * 8][nlo]      = (__bf16)p0;   // C-layout -> LDS 16x32
            plds[wave][v + half * 8][16 + nlo] = (__bf16)p1;
        }
        asm volatile("s_wait_dscnt 0" ::: "memory");   // same-wave LDS RAW

        // reload P as A-fragment (16x32)
        v16bf pa;
        {
            const __bf16* prow = &plds[wave][nlo][0];
#pragma unroll
            for (int v = 0; v < 8; ++v) {
                int ka = ((v & 4) << 2) + half * 8 + (v & 3) * 2;
                bf16x2 p = __builtin_bit_cast(bf16x2, *(const unsigned int*)(prow + ka));
                pa[2 * v] = p[0]; pa[2 * v + 1] = p[1];
            }
        }

        // O(16x64) += P(16x32) . V_tile(32x64) : 4 WMMAs; Vt gives contiguous K-pairs
#pragma unroll
        for (int t = 0; t < 4; ++t) {
            const __bf16* vcol = Vt + (size_t)(t * 16 + nlo) * SEQ + jb;
            v16bf b = loadB(vcol);
            v8f& ot = (t == 0) ? o0 : (t == 1) ? o1 : (t == 2) ? o2 : o3;
            ot = __builtin_amdgcn_wmma_f32_16x16x32_bf16(false, pa, false, b, (short)0, ot, false, false);
        }
    }

    // normalize and write [s][768]
#pragma unroll
    for (int v = 0; v < 8; ++v) {
        float inv = 1.0f / lrow[v];
        int r = qbase + v + half * 8;
        float* orow = out + (size_t)r * DIN + h * HD;
        orow[0 * 16 + nlo] = o0[v] * inv;
        orow[1 * 16 + nlo] = o1[v] * inv;
        orow[2 * 16 + nlo] = o2[v] * inv;
        orow[3 * 16 + nlo] = o3[v] * inv;
    }
}

// ---------------- launcher ----------------

extern "C" void kernel_launch(void* const* d_in, const int* in_sizes, int n_in,
                              void* d_out, int out_size, void* d_ws, size_t ws_size,
                              hipStream_t stream) {
    const float* x  = (const float*)d_in[0];
    const float* Wq = (const float*)d_in[1];
    const float* Wk = (const float*)d_in[2];
    const float* Wv = (const float*)d_in[3];
    float* out = (float*)d_out;

    char* ws = (char*)d_ws;
    __bf16* xb  = (__bf16*)ws;                                                   // 4096*768
    __bf16* Wt  = (__bf16*)(ws + (size_t)SEQ * DIN * 2);                         // 3*768*768
    __bf16* qkv = (__bf16*)(ws + (size_t)SEQ * DIN * 2 + (size_t)3 * DIN * DIN * 2); // 3*HSZ

    cvt_x_kernel<<<dim3(SEQ * DIN / 256), dim3(256), 0, stream>>>(x, xb);
    cvt_w_kernel<<<dim3(DIN * DIN / 256, 3), dim3(256), 0, stream>>>(Wq, Wk, Wv, Wt);
    proj_kernel<<<dim3((SEQ / 16) * (DIN / 128), 3), dim3(256), 0, stream>>>(xb, Wt, qkv);
    attn_kernel<<<dim3(SEQ / 16 / 8, NH), dim3(256), 0, stream>>>(qkv, out);
}
